// PPOAgent_7791070675942
// MI455X (gfx1250) — compile-verified, matrix operand order checked by reading, not run on hardware
//
#include <hip/hip_runtime.h>
#include <hip/hip_bf16.h>

// ---------------- problem constants ----------------
#define T_STEPS 524288
#define D_IN    64
#define H_DIM   256
#define A_DIM   18
#define GAMMA_C 0.99f
#define GL_C    (0.99f * 0.95f)
#define LN_EPS_C 1e-5f

#define HSTRIDE  264   // padded K-stride (halfwords) for 256-wide K arrays in LDS
#define W1STRIDE 72    // padded K-stride for K=64 weight
#define NBLK     512   // persistent workgroups for MLP kernels

typedef __attribute__((ext_vector_type(16))) _Float16 v16h;
typedef __attribute__((ext_vector_type(2)))  _Float16 h2;
typedef __attribute__((ext_vector_type(8)))  float    v8f;

#define WMMA_F16(a, b, c) \
  __builtin_amdgcn_wmma_f32_16x16x32_f16(false, (a), false, (b), (short)0, (c), false, false)

// K index of the first element held in VGPR r for lane-half h (16-bit A/B 16x32 layout,
// CDNA5 ISA 7.12.2): VGPR0..3 -> K pairs 0..7, VGPR4..7 -> K 16..23; half adds 8.
static __device__ __forceinline__ int kbase(int r, int h) {
  return ((r < 4) ? (2 * r) : (16 + 2 * (r - 4))) + 8 * h;
}

// A fragment (16x32 f16) built directly from global f32 rows (layer 1, K=64 input).
static __device__ __forceinline__ v16h frag_a_global(const float* __restrict__ X,
                                                     long row0, int kofs) {
  int lane = threadIdx.x & 31;
  int row = lane & 15, half = lane >> 4;
  const float* base = X + (row0 + row) * (long)D_IN;
  v16h a;
#pragma unroll
  for (int r = 0; r < 8; ++r) {
    int kb = kofs + kbase(r, half);
    a[2 * r]     = (_Float16)base[kb];
    a[2 * r + 1] = (_Float16)base[kb + 1];
  }
  return a;
}

// A fragment from the per-wave f16 H buffer in LDS ([16][HSTRIDE]).
static __device__ __forceinline__ v16h frag_a_lds(const _Float16* __restrict__ h, int kofs) {
  int lane = threadIdx.x & 31;
  int row = lane & 15, half = lane >> 4;
  v16h a;
#pragma unroll
  for (int r = 0; r < 8; ++r) {
    int kb = kofs + kbase(r, half);
    h2 p = *(const h2*)(h + row * HSTRIDE + kb);  // 4B-aligned ds_load_b32
    a[2 * r] = p[0];
    a[2 * r + 1] = p[1];
  }
  return a;
}

// B fragment (32x16 f16) from LDS weight stored column-major-by-N: w[n*stride + k].
static __device__ __forceinline__ v16h frag_b_lds(const _Float16* __restrict__ w,
                                                  int stride, int kofs, int nt) {
  int lane = threadIdx.x & 31;
  int n = (lane & 15) + nt * 16, half = lane >> 4;
  v16h b;
#pragma unroll
  for (int r = 0; r < 8; ++r) {
    int kb = kofs + kbase(r, half);
    h2 p = *(const h2*)(w + n * stride + kb);
    b[2 * r] = p[0];
    b[2 * r + 1] = p[1];
  }
  return b;
}

// 16-lane (half-wave) reductions: xor masks < 16 stay within each 16-lane half.
static __device__ __forceinline__ float hsum16(float x) {
  x += __shfl_xor(x, 1, 32);
  x += __shfl_xor(x, 2, 32);
  x += __shfl_xor(x, 4, 32);
  x += __shfl_xor(x, 8, 32);
  return x;
}
static __device__ __forceinline__ float hmax16(float x) {
  x = fmaxf(x, __shfl_xor(x, 1, 32));
  x = fmaxf(x, __shfl_xor(x, 2, 32));
  x = fmaxf(x, __shfl_xor(x, 4, 32));
  x = fmaxf(x, __shfl_xor(x, 8, 32));
  return x;
}

// LayerNorm(+affine)+ReLU over a full 16x256 activation held as 16 C-fragments,
// then f16 store into the per-wave LDS H buffer. C/D layout: lane half h holds
// rows v + 8h in VGPR v; column = 16*nt + (lane&15).
static __device__ __forceinline__ void ln_relu_store(v8f* acc,
                                                     const float* __restrict__ g,
                                                     const float* __restrict__ beta,
                                                     _Float16* __restrict__ myH) {
  int lane = threadIdx.x & 31;
  int col0 = lane & 15, half = lane >> 4;
#pragma unroll
  for (int v = 0; v < 8; ++v) {
    float s = 0.f, sq = 0.f;
#pragma unroll
    for (int nt = 0; nt < 16; ++nt) {
      float x = acc[nt][v];
      s += x;
      sq += x * x;
    }
    s = hsum16(s);
    sq = hsum16(sq);
    float mu  = s * (1.0f / 256.0f);
    float var = sq * (1.0f / 256.0f) - mu * mu;
    float rs  = rsqrtf(var + LN_EPS_C);
    int row = v + 8 * half;
#pragma unroll
    for (int nt = 0; nt < 16; ++nt) {
      int col = nt * 16 + col0;
      float y = (acc[nt][v] - mu) * rs * g[col] + beta[col];
      myH[row * HSTRIDE + col] = (_Float16)fmaxf(y, 0.0f);
    }
  }
}

// Shared trunk: X[16x64] -> LN/ReLU(256) -> LN/ReLU(256), result (f16) in myH.
static __device__ __forceinline__ void mlp_trunk(
    const float* __restrict__ X, long row0,
    const _Float16* __restrict__ w1T, const _Float16* __restrict__ w2T,
    const float* __restrict__ b1s, const float* __restrict__ g1s, const float* __restrict__ n1s,
    const float* __restrict__ b2s, const float* __restrict__ g2s, const float* __restrict__ n2s,
    _Float16* __restrict__ myH, v8f* acc) {
  int col0 = threadIdx.x & 15;
  // ----- layer 1: X @ W1 + b1 (bias preloaded into accumulators) -----
#pragma unroll
  for (int nt = 0; nt < 16; ++nt) {
    float bb = b1s[nt * 16 + col0];
#pragma unroll
    for (int e = 0; e < 8; ++e) acc[nt][e] = bb;
  }
#pragma unroll
  for (int kt = 0; kt < 2; ++kt) {
    v16h a = frag_a_global(X, row0, kt * 32);
#pragma unroll
    for (int nt = 0; nt < 16; ++nt) {
      v16h b = frag_b_lds(w1T, W1STRIDE, kt * 32, nt);
      acc[nt] = WMMA_F16(a, b, acc[nt]);
    }
  }
  ln_relu_store(acc, g1s, n1s, myH);
  // ----- layer 2: H1 @ W2 + b2 -----
#pragma unroll
  for (int nt = 0; nt < 16; ++nt) {
    float bb = b2s[nt * 16 + col0];
#pragma unroll
    for (int e = 0; e < 8; ++e) acc[nt][e] = bb;
  }
  for (int kt = 0; kt < 8; ++kt) {  // rolled: 16 WMMAs per k-step, A-frag reused 16x
    v16h a = frag_a_lds(myH, kt * 32);
#pragma unroll
    for (int nt = 0; nt < 16; ++nt) {
      v16h b = frag_b_lds(w2T, HSTRIDE, kt * 32, nt);
      acc[nt] = WMMA_F16(a, b, acc[nt]);
    }
  }
  ln_relu_store(acc, g2s, n2s, myH);  // all layer-2 A-frag reads precede these stores
}

// ---------------- actor: probs = softmax(MLP(states)) -> out[:,0:18] ----------------
__global__ void __launch_bounds__(256, 1)
actor_kernel(const float* __restrict__ states, float* __restrict__ out,
             const float* aw1, const float* ab1, const float* ag1, const float* an1,
             const float* aw2, const float* ab2, const float* ag2, const float* an2,
             const float* aw3, const float* ab3) {
  extern __shared__ char smem[];
  float* b1s = (float*)smem;
  float* g1s = b1s + 256;  float* n1s = g1s + 256;
  float* b2s = n1s + 256;  float* g2s = b2s + 256;  float* n2s = g2s + 256;
  float* b3s = n2s + 256;  // 32 floats
  _Float16* w1T = (_Float16*)(b3s + 32);
  _Float16* w2T = w1T + 256 * W1STRIDE;
  _Float16* w3T = w2T + 256 * HSTRIDE;   // 32 n-rows (cols >= 18 zero)
  _Float16* Hb  = w3T + 32 * HSTRIDE;    // 8 waves x 16 x HSTRIDE

  int tid = threadIdx.x;
  for (int i = tid; i < 64 * 256; i += 256) {      // W1: [k][n] f32 -> [n][k] f16
    int k = i >> 8, n = i & 255;
    w1T[n * W1STRIDE + k] = (_Float16)aw1[i];
  }
  for (int i = tid; i < 256 * 256; i += 256) {     // W2
    int k = i >> 8, n = i & 255;
    w2T[n * HSTRIDE + k] = (_Float16)aw2[i];
  }
  for (int i = tid; i < 256 * 32; i += 256) {      // W3 padded N=18 -> 32
    int k = i >> 5, n = i & 31;
    w3T[n * HSTRIDE + k] = (n < A_DIM) ? (_Float16)aw3[k * A_DIM + n] : (_Float16)0.0f;
  }
  for (int i = tid; i < 256; i += 256) {
    b1s[i] = ab1[i]; g1s[i] = ag1[i]; n1s[i] = an1[i];
    b2s[i] = ab2[i]; g2s[i] = ag2[i]; n2s[i] = an2[i];
  }
  if (tid < 32) b3s[tid] = (tid < A_DIM) ? ab3[tid] : 0.0f;
  __syncthreads();

  int wid = tid >> 5, lane = tid & 31;
  int col0 = lane & 15, half = lane >> 4;
  _Float16* myH = Hb + wid * (16 * HSTRIDE);
  int gw = blockIdx.x * 8 + wid;
  int nw = gridDim.x * 8;
  v8f acc[16];

  for (int tile = gw; tile < T_STEPS / 16; tile += nw) {
    long row0 = (long)tile * 16;
    if (tile + nw < T_STEPS / 16)
      __builtin_prefetch(states + (row0 + (long)nw * 16) * D_IN, 0, 3);

    mlp_trunk(states, row0, w1T, w2T, b1s, g1s, n1s, b2s, g2s, n2s, myH, acc);

    // head: logits[16x32] = H2 @ W3 + b3
    v8f l0, l1;
    float bb0 = b3s[col0], bb1 = b3s[16 + col0];
#pragma unroll
    for (int e = 0; e < 8; ++e) { l0[e] = bb0; l1[e] = bb1; }
    for (int kt = 0; kt < 8; ++kt) {
      v16h a  = frag_a_lds(myH, kt * 32);
      v16h b0 = frag_b_lds(w3T, HSTRIDE, kt * 32, 0);
      l0 = WMMA_F16(a, b0, l0);
      v16h b1 = frag_b_lds(w3T, HSTRIDE, kt * 32, 1);
      l1 = WMMA_F16(a, b1, l1);
    }
    // softmax over 18 columns per row, in registers
#pragma unroll
    for (int v = 0; v < 8; ++v) {
      float x0 = l0[v];
      float x1 = (col0 < (A_DIM - 16)) ? l1[v] : -3.0e38f;  // cols 16,17 valid
      float m  = hmax16(fmaxf(x0, x1));
      float e0 = __expf(x0 - m);
      float e1 = (col0 < (A_DIM - 16)) ? __expf(x1 - m) : 0.0f;
      float inv = 1.0f / hsum16(e0 + e1);
      float* orow = out + (row0 + v + 8 * half) * 20;
      orow[col0] = e0 * inv;
      if (col0 < (A_DIM - 16)) orow[16 + col0] = e1 * inv;
    }
  }
}

// ---------------- critic: values = MLP(X)[:,0] -> vals[T] ----------------
__global__ void __launch_bounds__(256, 1)
critic_kernel(const float* __restrict__ X, float* __restrict__ vals,
              const float* cw1, const float* cb1, const float* cg1, const float* cn1,
              const float* cw2, const float* cb2, const float* cg2, const float* cn2,
              const float* cw3, const float* cb3) {
  extern __shared__ char smem[];
  float* b1s = (float*)smem;
  float* g1s = b1s + 256;  float* n1s = g1s + 256;
  float* b2s = n1s + 256;  float* g2s = b2s + 256;  float* n2s = g2s + 256;
  _Float16* w1T = (_Float16*)(n2s + 256);
  _Float16* w2T = w1T + 256 * W1STRIDE;
  _Float16* w3T = w2T + 256 * HSTRIDE;   // 16 n-rows, only n==0 nonzero
  _Float16* Hb  = w3T + 16 * HSTRIDE;

  int tid = threadIdx.x;
  for (int i = tid; i < 64 * 256; i += 256) {
    int k = i >> 8, n = i & 255;
    w1T[n * W1STRIDE + k] = (_Float16)cw1[i];
  }
  for (int i = tid; i < 256 * 256; i += 256) {
    int k = i >> 8, n = i & 255;
    w2T[n * HSTRIDE + k] = (_Float16)cw2[i];
  }
  for (int i = tid; i < 256 * 16; i += 256) {
    int k = i >> 4, n = i & 15;
    w3T[n * HSTRIDE + k] = (n == 0) ? (_Float16)cw3[k] : (_Float16)0.0f;
  }
  for (int i = tid; i < 256; i += 256) {
    b1s[i] = cb1[i]; g1s[i] = cg1[i]; n1s[i] = cn1[i];
    b2s[i] = cb2[i]; g2s[i] = cg2[i]; n2s[i] = cn2[i];
  }
  __syncthreads();

  int wid = tid >> 5, lane = tid & 31;
  int col0 = lane & 15, half = lane >> 4;
  _Float16* myH = Hb + wid * (16 * HSTRIDE);
  int gw = blockIdx.x * 8 + wid;
  int nw = gridDim.x * 8;
  float bb3 = cb3[0];
  v8f acc[16];

  for (int tile = gw; tile < T_STEPS / 16; tile += nw) {
    long row0 = (long)tile * 16;
    if (tile + nw < T_STEPS / 16)
      __builtin_prefetch(X + (row0 + (long)nw * 16) * D_IN, 0, 3);

    mlp_trunk(X, row0, w1T, w2T, b1s, g1s, n1s, b2s, g2s, n2s, myH, acc);

    v8f l0;
#pragma unroll
    for (int e = 0; e < 8; ++e) l0[e] = bb3;
    for (int kt = 0; kt < 8; ++kt) {
      v16h a = frag_a_lds(myH, kt * 32);
      v16h b = frag_b_lds(w3T, HSTRIDE, kt * 32, 0);
      l0 = WMMA_F16(a, b, l0);
    }
    if (col0 == 0) {  // lanes 0 (rows 0-7) and 16 (rows 8-15) hold column N=0
#pragma unroll
      for (int v = 0; v < 8; ++v) vals[row0 + v + 8 * half] = l0[v];
    }
  }
}

// ---------------- GAE: blocked reverse linear-recurrence scan ----------------
// a[i] = d[i] + c[i]*a[i+1],  d = r + g*nv*m - v,  c = g*lam*m
#define CHUNK  4096
#define NCHUNK (T_STEPS / CHUNK)  // 128
#define PT     16                 // elements per thread (256 threads/chunk)

__global__ void gae_pass1(const float* __restrict__ rew, const float* __restrict__ msk,
                          const float* __restrict__ vals, const float* __restrict__ nvals,
                          float* __restrict__ blockA, float* __restrict__ blockP) {
  __shared__ float sA[256], sP[256];
  int tid = threadIdx.x, c = blockIdx.x;
  long i0 = (long)c * CHUNK + tid * PT;
  float a = 0.0f, p = 1.0f;
  for (int j = PT - 1; j >= 0; --j) {
    long i = i0 + j;
    float mk = msk[i];
    float cc = GL_C * mk;
    float dd = rew[i] + GAMMA_C * nvals[i] * mk - vals[i];
    a = dd + cc * a;
    p *= cc;
  }
  sA[tid] = a; sP[tid] = p;
  __syncthreads();
  if (tid == 0) {
    float ca = 0.0f, cp = 1.0f;
    for (int t = 255; t >= 0; --t) { ca = sA[t] + sP[t] * ca; cp *= sP[t]; }
    blockA[c] = ca;
    blockP[c] = cp;
  }
}

__global__ void gae_pass2(const float* __restrict__ blockA, const float* __restrict__ blockP,
                          float* __restrict__ blockC) {
  if (threadIdx.x == 0 && blockIdx.x == 0) {
    float carry = 0.0f;
    for (int k = NCHUNK - 1; k >= 0; --k) {
      blockC[k] = carry;                       // carry-in (adv just right of chunk k)
      carry = blockA[k] + blockP[k] * carry;
    }
  }
}

__global__ void gae_pass3(const float* __restrict__ rew, const float* __restrict__ msk,
                          const float* __restrict__ vals, const float* __restrict__ nvals,
                          const float* __restrict__ blockC, float* __restrict__ out) {
  __shared__ float sA[256], sP[256], sC[256];
  int tid = threadIdx.x, c = blockIdx.x;
  long i0 = (long)c * CHUNK + tid * PT;
  float a = 0.0f, p = 1.0f;
  for (int j = PT - 1; j >= 0; --j) {
    long i = i0 + j;
    float mk = msk[i];
    float cc = GL_C * mk;
    float dd = rew[i] + GAMMA_C * nvals[i] * mk - vals[i];
    a = dd + cc * a;
    p *= cc;
  }
  sA[tid] = a; sP[tid] = p;
  __syncthreads();
  if (tid == 0) {
    float carry = blockC[c];
    for (int t = 255; t >= 0; --t) {
      sC[t] = carry;
      carry = sA[t] + sP[t] * carry;
    }
  }
  __syncthreads();
  float av = sC[tid];
  for (int j = PT - 1; j >= 0; --j) {
    long i = i0 + j;
    float mk = msk[i];
    float cc = GL_C * mk;
    float dd = rew[i] + GAMMA_C * nvals[i] * mk - vals[i];
    av = dd + cc * av;
    out[i * 20 + 18] = av;                 // advantages
    out[i * 20 + 19] = av + vals[i];       // returns
  }
}

// ---------------- host launcher ----------------
extern "C" void kernel_launch(void* const* d_in, const int* in_sizes, int n_in,
                              void* d_out, int out_size, void* d_ws, size_t ws_size,
                              hipStream_t stream) {
  const float* states  = (const float*)d_in[0];
  const float* nstates = (const float*)d_in[1];
  const float* rewards = (const float*)d_in[2];
  const float* masks   = (const float*)d_in[3];
  const float* aw1 = (const float*)d_in[4],  *ab1 = (const float*)d_in[5];
  const float* ag1 = (const float*)d_in[6],  *an1 = (const float*)d_in[7];
  const float* aw2 = (const float*)d_in[8],  *ab2 = (const float*)d_in[9];
  const float* ag2 = (const float*)d_in[10], *an2 = (const float*)d_in[11];
  const float* aw3 = (const float*)d_in[12], *ab3 = (const float*)d_in[13];
  const float* cw1 = (const float*)d_in[14], *cb1 = (const float*)d_in[15];
  const float* cg1 = (const float*)d_in[16], *cn1 = (const float*)d_in[17];
  const float* cw2 = (const float*)d_in[18], *cb2 = (const float*)d_in[19];
  const float* cg2 = (const float*)d_in[20], *cn2 = (const float*)d_in[21];
  const float* cw3 = (const float*)d_in[22], *cb3 = (const float*)d_in[23];

  float* out = (float*)d_out;
  float* ws  = (float*)d_ws;
  float* vals  = ws;                       // T
  float* nvals = ws + T_STEPS;             // T
  float* bA = nvals + T_STEPS;             // NCHUNK
  float* bP = bA + NCHUNK;                 // NCHUNK
  float* bC = bP + NCHUNK;                 // NCHUNK

  size_t ldsActor = (size_t)(6 * 256 + 32) * sizeof(float) +
                    (size_t)(256 * W1STRIDE + 256 * HSTRIDE + 32 * HSTRIDE + 8 * 16 * HSTRIDE) * 2;
  size_t ldsCritic = (size_t)(6 * 256) * sizeof(float) +
                     (size_t)(256 * W1STRIDE + 256 * HSTRIDE + 16 * HSTRIDE + 8 * 16 * HSTRIDE) * 2;

  actor_kernel<<<NBLK, 256, ldsActor, stream>>>(states, out,
      aw1, ab1, ag1, an1, aw2, ab2, ag2, an2, aw3, ab3);
  critic_kernel<<<NBLK, 256, ldsCritic, stream>>>(states, vals,
      cw1, cb1, cg1, cn1, cw2, cb2, cg2, cn2, cw3, cb3);
  critic_kernel<<<NBLK, 256, ldsCritic, stream>>>(nstates, nvals,
      cw1, cb1, cg1, cn1, cw2, cb2, cg2, cn2, cw3, cb3);
  gae_pass1<<<NCHUNK, 256, 0, stream>>>(rewards, masks, vals, nvals, bA, bP);
  gae_pass2<<<1, 32, 0, stream>>>(bA, bP, bC);
  gae_pass3<<<NCHUNK, 256, 0, stream>>>(rewards, masks, vals, nvals, bC, out);
}
